// ElevationNet_38508676776022
// MI455X (gfx1250) — matrix operand order repaired and checked by reading, not compiled, not run on hardware
//
#include <hip/hip_runtime.h>
#include <hip/hip_bf16.h>

// ---------------------------------------------------------------------------
// ElevationNet forward for MI455X (gfx1250, wave32, WMMA).
// Dominant cost: 8x conv3x3 (576->576) fused into one implicit GEMM
//   M=4608 (head*co), N=3840 (n*h*w), K=5184 (ci*9), bf16 inputs, f32 accum,
//   v_wmma_f32_16x16x32_bf16. Branch-free 3-stage LDS pipeline via
//   global_load_async_to_lds_b128 + partial s_wait_asynccnt (in-order ASYNCcnt
//   completion), fused BN + SiLU epilogue. Packed operands (~88 MB) live in
//   the 192 MB L2, so the GEMM streams from L2 and is WMMA-throughput bound.
// ---------------------------------------------------------------------------

typedef __bf16 v16bf __attribute__((ext_vector_type(16)));
typedef __bf16 v8bf  __attribute__((ext_vector_type(8)));
typedef float  v8f   __attribute__((ext_vector_type(8)));

#define C_IN   576
#define NIMG   16
#define HF     12
#define WF     20
#define HW     (HF * WF)          // 240
#define KTOT   (C_IN * 9)         // 5184
#define MTOT   (8 * C_IN)         // 4608
#define NGEMM  (NIMG * HW)        // 3840
#define KSTEP  32
#define KSTEPS (KTOT / KSTEP)     // 162

__device__ __forceinline__ float sigm(float x) { return 1.0f / (1.0f + __expf(-x)); }
__device__ __forceinline__ float clampf(float x, float lo, float hi) { return fminf(fmaxf(x, lo), hi); }

// CDNA5 async global->LDS copy (16 bytes / lane), tracked by ASYNCcnt.
__device__ __forceinline__ void async_ld16(unsigned int lds_off, const void* gptr) {
    asm volatile("global_load_async_to_lds_b128 %0, %1, off"
                 :: "v"(lds_off), "v"((unsigned long long)(size_t)gptr)
                 : "memory");
}
// Wait until at most N async ops remain in flight (completion is in-order).
__device__ __forceinline__ void wait_async_le4() {
    asm volatile("s_wait_asynccnt 0x4" ::: "memory");
}
__device__ __forceinline__ void wait_async_le0() {
    asm volatile("s_wait_asynccnt 0x0" ::: "memory");
}

// --- BN fold: scale = g*rsqrt(v+eps), bias = b - m*scale  (m = head*576+co) --
__global__ void k_bnfold(const float* __restrict__ g, const float* __restrict__ b,
                         const float* __restrict__ m, const float* __restrict__ v,
                         float* __restrict__ scale, float* __restrict__ bias) {
    int i = blockIdx.x * blockDim.x + threadIdx.x;
    if (i < MTOT) {
        float s = g[i] * rsqrtf(v[i] + 1e-5f);
        scale[i] = s;
        bias[i]  = b[i] - m[i] * s;
    }
}

// --- weight pack: det_w1 flat is already [m][k] with k=ci*9+ky*3+kx ---------
__global__ void k_packW(const float* __restrict__ w1, __bf16* __restrict__ Wb) {
    int i = blockIdx.x * blockDim.x + threadIdx.x;
    if (i < MTOT * KTOT) Wb[i] = (__bf16)w1[i];
}

// --- im2col pack: Bm[j][k] = f[n][ci][y+ky-1][x+kx-1] (zero pad), j=n*240+hw -
__global__ void k_packB(const float* __restrict__ f, __bf16* __restrict__ Bm) {
    int i = blockIdx.x * blockDim.x + threadIdx.x;
    if (i >= NGEMM * KTOT) return;
    int j = i / KTOT, k = i - j * KTOT;
    int n = j / HW, hw = j - n * HW;
    int y = hw / WF, x = hw - y * WF;
    int ci = k / 9, r = k - ci * 9;
    int ky = r / 3, kx = r - ky * 3;
    int yy = y + ky - 1, xx = x + kx - 1;
    float val = 0.0f;
    if (yy >= 0 && yy < HF && xx >= 0 && xx < WF)
        val = f[((n * C_IN + ci) * HF + yy) * WF + xx];
    Bm[i] = (__bf16)val;
}

// --- main implicit-GEMM conv: C[m][j] = sum_k Wb[m][k]*Bm[j][k]; BN+SiLU ----
__global__ __launch_bounds__(256) void k_gemm(
    const __bf16* __restrict__ Wb, const __bf16* __restrict__ Bm,
    const float* __restrict__ scale, const float* __restrict__ bias,
    __bf16* __restrict__ Hm)
{
    __shared__ __bf16 As[3][128][KSTEP];   // [m_local][k]  8 KB per buffer
    __shared__ __bf16 Bs[3][128][KSTEP];   // [j_local][k]  (3-stage pipeline)
    const unsigned int BUFB = 128 * KSTEP * 2;   // 8192 B buffer stride

    const int tid   = threadIdx.x;
    const int ntile = blockIdx.x;          // 0..29
    const int mtile = blockIdx.y;          // 0..35
    const int row   = tid >> 1;            // 0..127
    const int half  = tid & 1;             // which 16-element K chunk
    const __bf16* aSrc = Wb + (long)(mtile * 128 + row) * KTOT + half * 16;
    const __bf16* bSrc = Bm + (long)(ntile * 128 + row) * KTOT + half * 16;
    // LDS byte offsets (flat->LDS truncates to addr[31:0])
    const unsigned int aDst = (unsigned int)(size_t)&As[0][row][half * 16];
    const unsigned int bDst = (unsigned int)(size_t)&Bs[0][row][half * 16];

    const int lane = tid & 31;
    const int wave = tid >> 5;             // 0..7
    const int wm = wave >> 2;              // 0..1 : 64-row band
    const int wn = wave & 3;               // 0..3 : 32-col band
    const int lr = lane & 15;
    const int hi = lane >> 4;              // lane half selects K sub-block

    v8f acc[4][2] = {};

    // one K-step of WMMA math from LDS buffer p
    auto compute = [&](int p) {
        // A fragment: 16x32 bf16. e<8 -> K = hi*8+e ; e>=8 -> K = 16+hi*8+(e-8)
        v16bf af[4];
        #pragma unroll
        for (int i = 0; i < 4; ++i) {
            const int r = wm * 64 + i * 16 + lr;
            const v8bf lo = *(const v8bf*)&As[p][r][hi * 8];
            const v8bf hp = *(const v8bf*)&As[p][r][16 + hi * 8];
            #pragma unroll
            for (int e = 0; e < 8; ++e) { af[i][e] = lo[e]; af[i][8 + e] = hp[e]; }
        }
        // B fragment: 32x16 bf16. e<8 -> K = hi*8+e ; e>=8 -> K = 16+hi*8+(e-8)
        v16bf bfr[2];
        #pragma unroll
        for (int j = 0; j < 2; ++j) {
            const int c = wn * 32 + j * 16 + lr;
            const v8bf blo = *(const v8bf*)&Bs[p][c][hi * 8];
            const v8bf bhp = *(const v8bf*)&Bs[p][c][16 + hi * 8];
            #pragma unroll
            for (int e = 0; e < 8; ++e) { bfr[j][e] = blo[e]; bfr[j][8 + e] = bhp[e]; }
        }
        #pragma unroll
        for (int i = 0; i < 4; ++i)
            #pragma unroll
            for (int j = 0; j < 2; ++j)
                acc[i][j] = __builtin_amdgcn_wmma_f32_16x16x32_bf16(
                    false, af[i], false, bfr[j], (short)0, acc[i][j], false, false);
    };

    // prologue: tiles 0 and 1 in flight; wait only for tile 0 (<=4 outstanding)
    async_ld16(aDst,             aSrc);
    async_ld16(aDst + 16,        aSrc + 8);
    async_ld16(bDst,             bSrc);
    async_ld16(bDst + 16,        bSrc + 8);
    async_ld16(aDst + BUFB,      aSrc + KSTEP);
    async_ld16(aDst + BUFB + 16, aSrc + KSTEP + 8);
    async_ld16(bDst + BUFB,      bSrc + KSTEP);
    async_ld16(bDst + BUFB + 16, bSrc + KSTEP + 8);
    wait_async_le4();
    __syncthreads();

    // steady state: branch-free; issue tile kb+2, compute tile kb, partial wait
    const __bf16* aNxt = aSrc + 2 * KSTEP;
    const __bf16* bNxt = bSrc + 2 * KSTEP;
    unsigned int wbuf = 2 * BUFB;   // write-buffer byte offset (tile kb+2)
    int p = 0;                      // compute-buffer index   (tile kb)
    for (int kb = 0; kb < KSTEPS - 2; ++kb) {
        async_ld16(aDst + wbuf,      aNxt);
        async_ld16(aDst + wbuf + 16, aNxt + 8);
        async_ld16(bDst + wbuf,      bNxt);
        async_ld16(bDst + wbuf + 16, bNxt + 8);
        aNxt += KSTEP;
        bNxt += KSTEP;
        wbuf = (wbuf == 2 * BUFB) ? 0u : wbuf + BUFB;

        compute(p);

        // tile kb+1 (issued last iteration) must have landed; ASYNC completion
        // is in-order, so <=4 outstanding leaves only tile kb+2 in flight.
        wait_async_le4();
        __syncthreads();
        p = (p == 2) ? 0 : p + 1;
    }
    // tail: tile KSTEPS-2 (wait for last tile), then tile KSTEPS-1
    compute(p);
    wait_async_le0();
    __syncthreads();
    p = (p == 2) ? 0 : p + 1;
    compute(p);

    // epilogue: C/D layout -> m_local = hi*8 + r, n_local = lr. BN + SiLU.
    #pragma unroll
    for (int i = 0; i < 4; ++i) {
        const int mb = mtile * 128 + wm * 64 + i * 16 + hi * 8;
        #pragma unroll
        for (int r = 0; r < 8; ++r) {
            const int m = mb + r;
            const float s = scale[m], bo = bias[m];
            #pragma unroll
            for (int j = 0; j < 2; ++j) {
                const int jc = ntile * 128 + wn * 32 + j * 16 + lr;
                float h  = acc[i][j][r] * s + bo;
                float si = h * sigm(h);
                Hm[m * NGEMM + jc] = (__bf16)si;
            }
        }
    }
}

// --- 14-channel projection: maps14[c][j] = b2[c] + sum_co w2[c][co]*h[...] ---
__global__ void k_proj(const __bf16* __restrict__ Hm, const float* __restrict__ w2,
                       const float* __restrict__ b2, float* __restrict__ maps14) {
    const int HEAD_OF[14] = {0,0,0,0,0,1,2,2,3,3,4,5,6,7};
    const int j   = blockIdx.x * blockDim.x + threadIdx.x;   // 0..3839
    const int c14 = blockIdx.y;                              // 0..13
    const int head = HEAD_OF[c14];
    float s = 0.0f;
    const __bf16* hp = Hm + (head * C_IN) * NGEMM + j;
    const float*  wp = w2 + c14 * C_IN;
    for (int co = 0; co < C_IN; ++co)
        s += wp[co] * (float)hp[co * NGEMM];
    maps14[c14 * NGEMM + j] = s + b2[c14];
}

// --- global-pool linear head -> single_out (N,6) -----------------------------
__global__ void k_single(const float* __restrict__ f, const float* __restrict__ linW,
                         const float* __restrict__ linb, float* __restrict__ out) {
    __shared__ float smean[C_IN];
    __shared__ float lin[10];
    const int n = blockIdx.x, tid = threadIdx.x;
    for (int c = tid; c < C_IN; c += blockDim.x) {
        const float* fp = f + ((long)n * C_IN + c) * HW;
        float s = 0.0f;
        for (int t = 0; t < HW; ++t) s += fp[t];
        smean[c] = s * (1.0f / HW);
    }
    __syncthreads();
    if (tid < 10) {
        float s = 0.0f;
        const float* wp = linW + tid * C_IN;
        for (int c = 0; c < C_IN; ++c) s += wp[c] * smean[c];
        lin[tid] = s + linb[tid];
    }
    __syncthreads();
    if (tid == 0) {
        const float eps = 1e-5f;
        float conf = clampf(sigm(lin[0]), eps, 1.0f - eps);
        int best = 0; float bv = lin[1];
        for (int i = 1; i < 5; ++i) if (lin[1 + i] > bv) { bv = lin[1 + i]; best = i; }
        out[n * 6 + 0] = conf;
        out[n * 6 + 1] = (float)best;
        out[n * 6 + 2] = sigm(lin[6]) * 3.0f;
        out[n * 6 + 3] = tanhf(lin[7]);
        out[n * 6 + 4] = clampf(sigm(lin[8]), eps, 1.0f - eps);
        out[n * 6 + 5] = clampf(sigm(lin[9]), eps, 1.0f - eps);
    }
}

// --- per-image decode: top-20 per class, box decode, stable sort by dist -----
__global__ void k_decode(const float* __restrict__ maps, const int* __restrict__ fwp,
                         const int* __restrict__ fhp, float* __restrict__ out) {
    __shared__ float sc[5][HW];
    __shared__ float rv[256];
    __shared__ int   ri[256];
    __shared__ float csc[100];
    __shared__ int   cix[100];
    __shared__ float dets[100][10];
    __shared__ int   order[100];

    const int n = blockIdx.x, tid = threadIdx.x;
    const float FW = (float)fwp[0], FH = (float)fhp[0];

    for (int t = tid; t < 5 * HW; t += blockDim.x) {
        int cls = t / HW, hw = t - cls * HW;
        int j = n * HW + hw;
        float cd = sigm(maps[5 * NGEMM + j]);
        float ht = sigm(maps[cls * NGEMM + j]);
        sc[cls][hw] = cd * ht;
    }
    __syncthreads();

    // top-20 per class, ties -> lowest index (matches lax.top_k)
    for (int cls = 0; cls < 5; ++cls) {
        for (int sel = 0; sel < 20; ++sel) {
            rv[tid] = (tid < HW) ? sc[cls][tid] : -1e30f;
            ri[tid] = tid;
            __syncthreads();
            for (int s = 128; s > 0; s >>= 1) {
                if (tid < s) {
                    float vo = rv[tid + s]; int io = ri[tid + s];
                    if (vo > rv[tid] || (vo == rv[tid] && io < ri[tid])) { rv[tid] = vo; ri[tid] = io; }
                }
                __syncthreads();
            }
            if (tid == 0) {
                int c = cls * 20 + sel;
                csc[c] = rv[0]; cix[c] = ri[0];
                sc[cls][ri[0]] = -1e30f;
            }
            __syncthreads();
        }
    }

    if (tid < 100) {
        const int hw = cix[tid];
        const int cls = tid / 20;
        const int j = n * HW + hw;
        const float gx = sigm(maps[6  * NGEMM + j]);
        const float gy = sigm(maps[7  * NGEMM + j]);
        const float gw = sigm(maps[8  * NGEMM + j]);
        const float gh = sigm(maps[9  * NGEMM + j]);
        const float gd = sigm(maps[10 * NGEMM + j]) * 3.0f;
        const float go = tanhf(maps[11 * NGEMM + j]);
        const float gv = sigm(maps[12 * NGEMM + j]);
        const float gs = sigm(maps[13 * NGEMM + j]);
        const float xi = (float)(hw % WF), yi = (float)(hw / WF);
        const float sx = FW / WF, sy = FH / HF;
        const float cx = clampf(xi + gx, 0.0f, WF - 1.0f) * sx;
        const float cy = clampf(yi + gy, 0.0f, HF - 1.0f) * sy;
        const float bw = clampf(gw, 0.0f, 1.0f) * FW;
        const float bh = clampf(gh, 0.0f, 1.0f) * FH;
        dets[tid][0] = csc[tid];
        dets[tid][1] = (float)cls;
        dets[tid][2] = gd;
        dets[tid][3] = go;
        dets[tid][4] = gv;
        dets[tid][5] = gs;
        dets[tid][6] = clampf(cx - bw * 0.5f, 0.0f, FW - 1.0f);
        dets[tid][7] = clampf(cy - bh * 0.5f, 0.0f, FH - 1.0f);
        dets[tid][8] = clampf(cx + bw * 0.5f, 0.0f, FW - 1.0f);
        dets[tid][9] = clampf(cy + bh * 0.5f, 0.0f, FH - 1.0f);
    }
    __syncthreads();

    if (tid == 0) {   // stable insertion argsort by dets[:,2] ascending
        for (int i = 0; i < 100; ++i) order[i] = i;
        for (int i = 1; i < 100; ++i) {
            int key = order[i];
            float kd = dets[key][2];
            int jj = i - 1;
            while (jj >= 0 && dets[order[jj]][2] > kd) { order[jj + 1] = order[jj]; --jj; }
            order[jj + 1] = key;
        }
    }
    __syncthreads();

    if (tid < 200) {
        int r = tid / 10, c = tid - r * 10;
        out[96 + n * 200 + tid] = dets[order[r]][c];
    }
}

// ---------------------------------------------------------------------------
extern "C" void kernel_launch(void* const* d_in, const int* in_sizes, int n_in,
                              void* d_out, int out_size, void* d_ws, size_t ws_size,
                              hipStream_t stream) {
    const float* f    = (const float*)d_in[0];
    const float* w1   = (const float*)d_in[1];
    const float* bng  = (const float*)d_in[2];
    const float* bnb  = (const float*)d_in[3];
    const float* bnm  = (const float*)d_in[4];
    const float* bnv  = (const float*)d_in[5];
    const float* w2   = (const float*)d_in[6];
    const float* b2   = (const float*)d_in[7];
    const float* linW = (const float*)d_in[8];
    const float* linb = (const float*)d_in[9];
    const int*   fwp  = (const int*)d_in[10];
    const int*   fhp  = (const int*)d_in[11];
    float* out = (float*)d_out;

    char* ws = (char*)d_ws;
    const size_t offWb  = 0;                                   // 4608*5184*2
    const size_t offBm  = offWb + (size_t)MTOT * KTOT * 2;     // +47,775,744
    const size_t offHm  = offBm + (size_t)NGEMM * KTOT * 2;    // +39,813,120
    const size_t offSc  = offHm + (size_t)MTOT * NGEMM * 2;    // +35,389,440
    const size_t offBi  = offSc + (size_t)MTOT * 4;
    const size_t offMp  = offBi + (size_t)MTOT * 4;
    __bf16* Wb   = (__bf16*)(ws + offWb);
    __bf16* Bm   = (__bf16*)(ws + offBm);
    __bf16* Hm   = (__bf16*)(ws + offHm);
    float* scale = (float*)(ws + offSc);
    float* bias  = (float*)(ws + offBi);
    float* maps  = (float*)(ws + offMp);

    k_bnfold<<<(MTOT + 255) / 256, 256, 0, stream>>>(bng, bnb, bnm, bnv, scale, bias);
    k_packW <<<(MTOT * KTOT + 255) / 256, 256, 0, stream>>>(w1, Wb);
    k_packB <<<(NGEMM * KTOT + 255) / 256, 256, 0, stream>>>(f, Bm);
    k_gemm  <<<dim3(NGEMM / 128, MTOT / 128), 256, 0, stream>>>(Wb, Bm, scale, bias, Hm);
    k_proj  <<<dim3(NGEMM / 256, 14), 256, 0, stream>>>(Hm, w2, b2, maps);
    k_single<<<NIMG, 256, 0, stream>>>(f, linW, linb, out);
    k_decode<<<NIMG, 256, 0, stream>>>(maps, fwp, fhp, out);
}
